// SiameseModel_58755152609647
// MI455X (gfx1250) — matrix operand order repaired and checked by reading, not compiled, not run on hardware
//
#include <hip/hip_runtime.h>

#define BATCH  256
#define MAXLEN 128
#define EMB    300
#define EPAD   320          // EMB padded to multiple of 32
#define HID    1024
#define GATES  4096
#define KPAD   1344         // EPAD + HID
#define KCH    (KPAD / 32)  // 42 k-chunks
#define BB2    512          // fused batch (both sequences)
#define M2D    512
#define HP     1025         // HID + 1 (len feature)
#define HC     3332         // 3*HP + 1 + BATCH

typedef __attribute__((ext_vector_type(16))) __bf16 v16bf;
typedef __attribute__((ext_vector_type(8)))  float  v8f;

union FragU { uint4 q[2]; v16bf v; };

__device__ __forceinline__ __bf16 f2bf(float f) {
    unsigned u = __builtin_bit_cast(unsigned, f);
    u = (u + 0x7FFFu + ((u >> 16) & 1u)) >> 16;       // round-to-nearest-even
    unsigned short s = (unsigned short)u;
    return __builtin_bit_cast(__bf16, s);
}

__device__ __forceinline__ float sigf(float x) { return 1.0f / (1.0f + expf(-x)); }

__device__ __forceinline__ v8f wmma_bf16(const FragU& a, const FragU& b, v8f c) {
    return __builtin_amdgcn_wmma_f32_16x16x32_bf16(false, a.v, false, b.v, (short)0, c,
                                                   false, false);
}

// ---------------------------------------------------------------------------
// Prep: gather embeddings (bf16, [t][b][EPAD], zero-padded E 300->320)
// ---------------------------------------------------------------------------
__global__ void embed_gather(const int* __restrict__ input1,
                             const int* __restrict__ input2,
                             const float* __restrict__ emb,
                             __bf16* __restrict__ xemb) {
    int b = blockIdx.x, t = blockIdx.y, e = threadIdx.x;   // block = 320 threads
    int tok = (b < BATCH) ? input1[b * MAXLEN + t] : input2[(b - BATCH) * MAXLEN + t];
    float v = (e < EMB) ? emb[(size_t)tok * EMB + e] : 0.0f;
    xemb[((size_t)t * BB2 + b) * EPAD + e] = f2bf(v);
}

// ---------------------------------------------------------------------------
// Prep: transpose + pad lstm_kernel (1324 x 4096) -> WkT[n][KPAD] bf16.
// Rows 0..299 = Wx, 300..319 = zero pad, 320..1343 = Wh.
// ---------------------------------------------------------------------------
__global__ void wkt_prep(const float* __restrict__ kern, __bf16* __restrict__ wkt) {
    int n = blockIdx.x;
    for (int k = threadIdx.x; k < KPAD; k += 256) {
        float v = 0.0f;
        if (k < EMB)        v = kern[(size_t)k * GATES + n];
        else if (k >= EPAD) v = kern[(size_t)(k - (EPAD - EMB)) * GATES + n];
        wkt[(size_t)n * KPAD + k] = f2bf(v);
    }
}

// ---------------------------------------------------------------------------
// Prep: concat initial c/h for both sequences; seed bf16 h ping buffer.
// ---------------------------------------------------------------------------
__global__ void init_state(const float* __restrict__ c1, const float* __restrict__ c2,
                           const float* __restrict__ h1, const float* __restrict__ h2,
                           float* __restrict__ cbuf, float* __restrict__ hf32,
                           __bf16* __restrict__ hbf0) {
    int b = blockIdx.x;
    const float* cs = (b < BATCH) ? (c1 + (size_t)b * HID) : (c2 + (size_t)(b - BATCH) * HID);
    const float* hs = (b < BATCH) ? (h1 + (size_t)b * HID) : (h2 + (size_t)(b - BATCH) * HID);
    for (int n = threadIdx.x; n < HID; n += 256) {
        cbuf[(size_t)b * HID + n] = cs[n];
        float h = hs[n];
        hf32[(size_t)b * HID + n] = h;
        hbf0[(size_t)b * HID + n] = f2bf(h);
    }
}

// ---------------------------------------------------------------------------
// Fused LSTM step. Wave tile: M=32 (two 16-row A fragments) x N=16 x 4 gates
// -> 8 accumulators, 8 WMMAs per 32-deep k-chunk. Each B fragment is reused
// by two A fragments, halving L2 weight traffic vs a 16x16 tile. Loads are
// left to the scheduler to pipeline; __launch_bounds__(256,1) gives the
// backend a full register budget so nothing spills to scratch.
// Grid: 128 blocks x 256 threads = 1024 waves = (512/32) x (1024/16).
// ---------------------------------------------------------------------------
__global__ void __launch_bounds__(256, 1)
lstm_step_kernel(const __bf16* __restrict__ xemb,
                 const __bf16* __restrict__ wkt,
                 const __bf16* __restrict__ hcur,
                 __bf16* __restrict__ hnext,
                 float* __restrict__ cbuf,
                 float* __restrict__ hf32,
                 const float* __restrict__ bias,
                 const int* __restrict__ seqlen1,
                 const int* __restrict__ seqlen2,
                 int t) {
    const int lane = threadIdx.x & 31;
    const int wave = (blockIdx.x << 3) + (threadIdx.x >> 5);
    const int m0 = (wave & 15) << 5;     // 16 M-tiles of 32 rows
    const int n0 = (wave >> 4) << 4;     // 64 N-tiles of 16 h-columns

    // A fragments (k order chosen here, same as B): lane row = lane&15,
    // lanes 0-15 hold K 0-15, lanes 16-31 hold K 16-31 (contiguous in k).
    const int arow = m0 + (lane & 15);
    const int akb  = (lane >> 4) << 4;
    const __bf16* aX0 = xemb + ((size_t)t * BB2 + arow) * EPAD + akb;
    const __bf16* aX1 = aX0 + (size_t)16 * EPAD;
    const __bf16* aH0 = hcur + (size_t)arow * HID + akb;
    const __bf16* aH1 = aH0 + (size_t)16 * HID;

    // B fragments from transposed weights WkT[n][k]: lane column = lane&15,
    // lanes 0-15 hold K 0-15, lanes 16-31 hold K 16-31 (contiguous in k).
    const int bcol = n0 + (lane & 15);
    const int bkb  = (lane >> 4) << 4;
    const __bf16* bp0 = wkt + (size_t)(bcol          ) * KPAD + bkb;   // i
    const __bf16* bp1 = wkt + (size_t)(bcol + HID    ) * KPAD + bkb;   // j
    const __bf16* bp2 = wkt + (size_t)(bcol + 2 * HID) * KPAD + bkb;   // f
    const __bf16* bp3 = wkt + (size_t)(bcol + 3 * HID) * KPAD + bkb;   // o

    // Warm L2/L0 for the weight rows (global_prefetch_b8 path on gfx1250).
    __builtin_prefetch(bp0, 0, 3);
    __builtin_prefetch(bp1, 0, 3);
    __builtin_prefetch(bp2, 0, 3);
    __builtin_prefetch(bp3, 0, 3);

    v8f acc[2][4] = {};

    for (int kc = 0; kc < KCH; ++kc) {
        const int kb = kc << 5;
        FragU a0, a1, b0, b1, b2, b3;
        const __bf16* a0s = (kc < EPAD / 32) ? (aX0 + kb) : (aH0 + (kb - EPAD));
        const __bf16* a1s = (kc < EPAD / 32) ? (aX1 + kb) : (aH1 + (kb - EPAD));
        a0.q[0] = *(const uint4*)(a0s);       a0.q[1] = *(const uint4*)(a0s + 8);
        a1.q[0] = *(const uint4*)(a1s);       a1.q[1] = *(const uint4*)(a1s + 8);
        b0.q[0] = *(const uint4*)(bp0 + kb);  b0.q[1] = *(const uint4*)(bp0 + kb + 8);
        b1.q[0] = *(const uint4*)(bp1 + kb);  b1.q[1] = *(const uint4*)(bp1 + kb + 8);
        b2.q[0] = *(const uint4*)(bp2 + kb);  b2.q[1] = *(const uint4*)(bp2 + kb + 8);
        b3.q[0] = *(const uint4*)(bp3 + kb);  b3.q[1] = *(const uint4*)(bp3 + kb + 8);
        acc[0][0] = wmma_bf16(a0, b0, acc[0][0]);
        acc[1][0] = wmma_bf16(a1, b0, acc[1][0]);
        acc[0][1] = wmma_bf16(a0, b1, acc[0][1]);
        acc[1][1] = wmma_bf16(a1, b1, acc[1][1]);
        acc[0][2] = wmma_bf16(a0, b2, acc[0][2]);
        acc[1][2] = wmma_bf16(a1, b2, acc[1][2]);
        acc[0][3] = wmma_bf16(a0, b3, acc[0][3]);
        acc[1][3] = wmma_bf16(a1, b3, acc[1][3]);
    }

    // Epilogue. C/D layout: VGPR r, lanes 0-15 -> M=r, lanes 16-31 -> M=8+r; N=lane&15.
    const int gcol  = n0 + (lane & 15);
    const int rbase = (lane >> 4) << 3;
    const float bi  = bias[gcol];
    const float bj  = bias[HID + gcol];
    const float bff = bias[2 * HID + gcol];
    const float bo  = bias[3 * HID + gcol];
    for (int rt = 0; rt < 2; ++rt) {
        for (int r = 0; r < 8; ++r) {
            int row = m0 + (rt << 4) + rbase + r;
            int sl = (row < BATCH) ? seqlen1[row] : seqlen2[row - BATCH];
            size_t idx = (size_t)row * HID + gcol;
            float gi = acc[rt][0][r] + bi;
            float gj = acc[rt][1][r] + bj;
            float gf = acc[rt][2][r] + bff;
            float go = acc[rt][3][r] + bo;
            float cold = cbuf[idx];
            float cn = sigf(gf + 1.0f) * cold + sigf(gi) * tanhf(gj);
            float hn = tanhf(cn) * sigf(go);
            bool valid = (t < sl);
            float cw = valid ? cn : cold;
            float hw = valid ? hn : hf32[idx];
            cbuf[idx] = cw;
            hf32[idx] = hw;
            hnext[idx] = f2bf(hw);
        }
    }
}

// ---------------------------------------------------------------------------
// Tail (fp32 exact): features, interaction, MLP.
// hc layout per row b: [h1e(1025) | h2e(1025) | dist(1) | mul(1025) | inter(256)]
// ---------------------------------------------------------------------------
__global__ void feat_kernel(const float* __restrict__ hf32,
                            const int* __restrict__ s1, const int* __restrict__ s2,
                            float* __restrict__ hc) {
    __shared__ float red[256];
    int b = blockIdx.x;
    float l1 = (float)s1[b] / (float)MAXLEN;
    float l2 = (float)s2[b] / (float)MAXLEN;
    float acc = 0.0f;
    for (int j = threadIdx.x; j < HP; j += 256) {
        float v1 = (j < HID) ? hf32[(size_t)b * HID + j] : l1;
        float v2 = (j < HID) ? hf32[(size_t)(b + BATCH) * HID + j] : l2;
        hc[(size_t)b * HC + j] = v1;
        hc[(size_t)b * HC + HP + j] = v2;
        hc[(size_t)b * HC + 2 * HP + 1 + j] = v1 * v2;
        float d = v1 - v2;
        acc += d * d;
    }
    red[threadIdx.x] = acc;
    __syncthreads();
    for (int s = 128; s > 0; s >>= 1) {
        if ((int)threadIdx.x < s) red[threadIdx.x] += red[threadIdx.x + s];
        __syncthreads();
    }
    if (threadIdx.x == 0) hc[(size_t)b * HC + 2 * HP] = red[0];
}

__global__ void gemm_tmp(const float* __restrict__ hc, const float* __restrict__ W_h,
                         float* __restrict__ tmp) {
    int n = blockIdx.x * 256 + threadIdx.x;
    int b = blockIdx.y;
    if (n >= HP) return;
    float acc = 0.0f;
    const float* a = hc + (size_t)b * HC;
    for (int k = 0; k < HP; ++k) acc += a[k] * W_h[(size_t)k * HP + n];
    tmp[(size_t)b * HP + n] = acc;
}

__global__ void gemm_inter(const float* __restrict__ tmp, float* hc) {
    int b2 = threadIdx.x;
    int b = blockIdx.y;
    float acc = 0.0f;
    const float* a  = tmp + (size_t)b * HP;
    const float* h2 = hc + (size_t)b2 * HC + HP;
    for (int k = 0; k < HP; ++k) acc += a[k] * h2[k];
    hc[(size_t)b * HC + 3 * HP + 1 + b2] = acc;
}

__global__ void gemm_e1(const float* __restrict__ hc, const float* __restrict__ W1,
                        const float* __restrict__ b1, float* __restrict__ e1) {
    int n = blockIdx.x * 256 + threadIdx.x;
    int b = blockIdx.y;
    float acc = b1[n];
    const float* a = hc + (size_t)b * HC;
    for (int k = 0; k < HC; ++k) acc += a[k] * W1[(size_t)k * M2D + n];
    e1[(size_t)b * M2D + n] = fmaxf(acc, 0.0f);
}

__global__ void gemm_out(const float* __restrict__ e1, const float* __restrict__ W2,
                         const float* __restrict__ b2, float* __restrict__ out) {
    int tid = blockIdx.x * 256 + threadIdx.x;
    if (tid >= BATCH * 2) return;
    int b = tid >> 1, j = tid & 1;
    float acc = b2[j];
    const float* a = e1 + (size_t)b * M2D;
    for (int k = 0; k < M2D; ++k) acc += a[k] * W2[(size_t)k * 2 + j];
    out[(size_t)b * 2 + j] = acc;
}

// ---------------------------------------------------------------------------
extern "C" void kernel_launch(void* const* d_in, const int* in_sizes, int n_in,
                              void* d_out, int out_size, void* d_ws, size_t ws_size,
                              hipStream_t stream) {
    const int*   input1      = (const int*)  d_in[0];
    const int*   input2      = (const int*)  d_in[1];
    const int*   seqlen1     = (const int*)  d_in[2];
    const int*   seqlen2     = (const int*)  d_in[3];
    const float* embeddings  = (const float*)d_in[4];
    const float* lstm_kernel = (const float*)d_in[5];
    const float* lstm_bias   = (const float*)d_in[6];
    const float* c1_init     = (const float*)d_in[7];
    const float* h1_init     = (const float*)d_in[8];
    const float* c2_init     = (const float*)d_in[9];
    const float* h2_init     = (const float*)d_in[10];
    const float* W_h         = (const float*)d_in[11];
    const float* W1          = (const float*)d_in[12];
    const float* b1          = (const float*)d_in[13];
    const float* W2          = (const float*)d_in[14];
    const float* b2v         = (const float*)d_in[15];
    float* out = (float*)d_out;

    char* ws = (char*)d_ws;
    size_t off = 0;
    auto alloc = [&](size_t bytes) -> void* {
        void* p = ws + off;
        off = (off + bytes + 255) & ~(size_t)255;
        return p;
    };

    __bf16* xemb = (__bf16*)alloc((size_t)MAXLEN * BB2 * EPAD * sizeof(__bf16));
    __bf16* wkt  = (__bf16*)alloc((size_t)GATES * KPAD * sizeof(__bf16));
    __bf16* hbf0 = (__bf16*)alloc((size_t)BB2 * HID * sizeof(__bf16));
    __bf16* hbf1 = (__bf16*)alloc((size_t)BB2 * HID * sizeof(__bf16));
    float*  cbuf = (float*)alloc((size_t)BB2 * HID * sizeof(float));
    float*  hf32 = (float*)alloc((size_t)BB2 * HID * sizeof(float));
    float*  hc   = (float*)alloc((size_t)BATCH * HC * sizeof(float));
    float*  tmp  = (float*)alloc((size_t)BATCH * HP * sizeof(float));
    float*  e1   = (float*)alloc((size_t)BATCH * M2D * sizeof(float));

    embed_gather<<<dim3(BB2, MAXLEN), 320, 0, stream>>>(input1, input2, embeddings, xemb);
    wkt_prep<<<GATES, 256, 0, stream>>>(lstm_kernel, wkt);
    init_state<<<BB2, 256, 0, stream>>>(c1_init, c2_init, h1_init, h2_init, cbuf, hf32, hbf0);

    __bf16* hb[2] = {hbf0, hbf1};
    for (int t = 0; t < MAXLEN; ++t) {
        lstm_step_kernel<<<128, 256, 0, stream>>>(xemb, wkt, hb[t & 1], hb[(t + 1) & 1],
                                                  cbuf, hf32, lstm_bias, seqlen1, seqlen2, t);
    }

    feat_kernel<<<BATCH, 256, 0, stream>>>(hf32, seqlen1, seqlen2, hc);
    gemm_tmp<<<dim3((HP + 255) / 256, BATCH), 256, 0, stream>>>(hc, W_h, tmp);
    gemm_inter<<<dim3(1, BATCH), 256, 0, stream>>>(tmp, hc);
    gemm_e1<<<dim3(M2D / 256, BATCH), 256, 0, stream>>>(hc, W1, b1, e1);
    gemm_out<<<2, 256, 0, stream>>>(e1, W2, b2v, out);
}
